// TerroirGNN_31945966748099
// MI455X (gfx1250) — matrix operand order repaired and checked by reading, not compiled, not run on hardware
//
#include <hip/hip_runtime.h>
#include <hip/hip_bf16.h>
#include <stdint.h>

#define N_NODES 50000
#define N_EDGES 800000
#define N_GRAPH 128
#define C_IN    128
#define C_HID   64
#define C_OUT   32
#define N_HEAD  4
#define C_GAT   256                 /* N_HEAD * C_HID */
#define ETOT    (N_EDGES + N_NODES) /* self-loops appended */
#define BN_EPS  1e-5f

typedef __attribute__((ext_vector_type(16))) __bf16 v16bf;
typedef __attribute__((ext_vector_type(8)))  float  v8f;

__device__ __forceinline__ unsigned short f2bf(float f) {
  unsigned int u = __float_as_uint(f);
  u += 0x7fffu + ((u >> 16) & 1u);          // round-to-nearest-even
  return (unsigned short)(u >> 16);
}

__device__ __forceinline__ void atomicMaxF(float* a, float v) {
  // signed-max for positives, unsigned-min for negatives: race-safe ordered-float trick
  if (__float_as_int(v) >= 0) atomicMax((int*)a, __float_as_int(v));
  else                        atomicMin((unsigned int*)a, __float_as_uint(v));
}

// ---------------- fill / BN-input ----------------
__global__ void k_fill(float* p, float v, int n) {
  int i = blockIdx.x * blockDim.x + threadIdx.x;
  if (i < n) p[i] = v;
}

__global__ void k_bn_in(const float* __restrict__ x, const float* __restrict__ g,
                        const float* __restrict__ b, const float* __restrict__ mn,
                        const float* __restrict__ vr, unsigned short* __restrict__ xb) {
  int i = blockIdx.x * blockDim.x + threadIdx.x;
  if (i >= N_NODES * C_IN) return;
  int c = i & (C_IN - 1);
  float v = (x[i] - mn[c]) * rsqrtf(vr[c] + BN_EPS) * g[c] + b[c];
  xb[i] = f2bf(v);
}

// ---------------- weight pack: [K,Ko] f32 -> bf16 B-fragment layout ----------------
// packed index = (((kt*(Ko/16) + nt)*32 + lane)*16 + j)
// lane 0-15 : col = nt*16 + lane,      K = kt*32 + {0..7,16..23}[j]
// lane 16-31: col = nt*16 + (lane-16), K = kt*32 + {8..15,24..31}[j]
__global__ void k_packW(const float* __restrict__ W, unsigned short* __restrict__ Bp,
                        int K, int Ko) {
  int idx = blockIdx.x * blockDim.x + threadIdx.x;
  if (idx >= K * Ko) return;
  int j    = idx & 15;
  int lane = (idx >> 4) & 31;
  int t    = idx >> 9;
  int ntn  = Ko >> 4;
  int kt   = t / ntn, nt = t % ntn;
  int col  = (nt << 4) + (lane & 15);
  int krow = (kt << 5) + ((lane >> 4) << 3) + (j & 7) + ((j >> 3) << 4);
  Bp[idx] = f2bf(W[krow * Ko + col]);
}

// ---------------- bf16 WMMA GEMM: C[M,Ko] = A[M,K] * B[K,Ko] ----------------
// One wave produces a 16x64 strip: 4 accumulators, A fragment loaded once per
// K-step and reused across 4 back-to-back v_wmma_f32_16x16x32_bf16.
__global__ void k_gemm_bf16(const unsigned short* __restrict__ A,
                            const unsigned short* __restrict__ Bp,
                            float* __restrict__ C, int M, int K, int Ko) {
  int wave = (int)((blockIdx.x * blockDim.x + threadIdx.x) >> 5);
  int lane = threadIdx.x & 31;
  int ntn  = Ko >> 4;         // 16x16 tiles along N
  int ngrp = Ko >> 6;         // 16x64 strips along N
  int strips = (M >> 4) * ngrp;
  if (wave >= strips) return;
  int mt  = wave / ngrp;
  int nt0 = (wave % ngrp) << 2;            // first 16-wide tile of this strip
  int row  = (mt << 4) + (lane & 15);
  int half = lane >> 4;

  v8f acc0 = {}, acc1 = {}, acc2 = {}, acc3 = {};

  // per-lane pointers; constant strides -> compiler folds into offsets
  const unsigned short* ap = A + (size_t)row * K + half * 8;
  const unsigned short* bp = Bp + (((size_t)nt0 * 32 + lane) << 4);
  const size_t bstride = (size_t)ntn << 9;     // elements per K-step in packed B

  for (int kt = 0; kt < (K >> 5); ++kt) {
    // A fragment: lanes 0-15 hold K {0..7,16..23}, lanes 16-31 hold K {8..15,24..31}
    union { uint4 q[2]; v16bf v; } ua;
    ua.q[0] = *(const uint4*)(ap);
    ua.q[1] = *(const uint4*)(ap + 16);
    // 4 pre-packed B fragments, contiguous per lane (each = 2 x b128)
    union { uint4 q[2]; v16bf v; } ub0, ub1, ub2, ub3;
    ub0.q[0] = *(const uint4*)(bp);        ub0.q[1] = *(const uint4*)(bp + 8);
    ub1.q[0] = *(const uint4*)(bp + 512);  ub1.q[1] = *(const uint4*)(bp + 520);
    ub2.q[0] = *(const uint4*)(bp + 1024); ub2.q[1] = *(const uint4*)(bp + 1032);
    ub3.q[0] = *(const uint4*)(bp + 1536); ub3.q[1] = *(const uint4*)(bp + 1544);
    acc0 = __builtin_amdgcn_wmma_f32_16x16x32_bf16(false, ua.v, false, ub0.v, (short)0, acc0, false, false);
    acc1 = __builtin_amdgcn_wmma_f32_16x16x32_bf16(false, ua.v, false, ub1.v, (short)0, acc1, false, false);
    acc2 = __builtin_amdgcn_wmma_f32_16x16x32_bf16(false, ua.v, false, ub2.v, (short)0, acc2, false, false);
    acc3 = __builtin_amdgcn_wmma_f32_16x16x32_bf16(false, ua.v, false, ub3.v, (short)0, acc3, false, false);
    ap += 32;
    bp += bstride;
  }

  // C/D layout: VGPR v, lanes 0-15 -> row mt*16+v, lanes 16-31 -> row mt*16+8+v; col = lane&15
  int rbase = (mt << 4) + half * 8;
  int col   = (nt0 << 4) + (lane & 15);
  float* c0 = C + (size_t)rbase * Ko + col;
#pragma unroll
  for (int v = 0; v < 8; ++v) {
    float* cr = c0 + (size_t)v * Ko;
    cr[0]  = acc0[v];
    cr[16] = acc1[v];
    cr[32] = acc2[v];
    cr[48] = acc3[v];
  }
}

// ---------------- attention projections a_src/a_dst [N, H] ----------------
__global__ void k_att(const float* __restrict__ h, const float* __restrict__ as_,
                      const float* __restrict__ ad_, float* __restrict__ asrc,
                      float* __restrict__ adst) {
  int i = blockIdx.x * blockDim.x + threadIdx.x;
  if (i >= N_NODES * N_HEAD) return;
  int hd = i & (N_HEAD - 1);
  int n  = i >> 2;
  const float* hp = h + (size_t)n * C_GAT + hd * C_HID;
  float ss = 0.f, sd = 0.f;
#pragma unroll 8
  for (int c = 0; c < C_HID; ++c) {
    float v = hp[c];
    ss += v * as_[hd * C_HID + c];
    sd += v * ad_[hd * C_HID + c];
  }
  asrc[i] = ss;
  adst[i] = sd;
}

__device__ __forceinline__ void edge_nodes(const int* ei, int e, int& s, int& d) {
  if (e < N_EDGES) { s = ei[e]; d = ei[N_EDGES + e]; }
  else             { s = e - N_EDGES; d = e - N_EDGES; }   // self-loop
}

// ---------------- segment softmax pass 1: per-(dst,head) max ----------------
__global__ void k_edge_max(const int* __restrict__ ei, const float* __restrict__ asrc,
                           const float* __restrict__ adst, float* __restrict__ m) {
  int e = blockIdx.x * blockDim.x + threadIdx.x;
  if (e >= ETOT) return;
  int sn, dn; edge_nodes(ei, e, sn, dn);
#pragma unroll
  for (int hd = 0; hd < N_HEAD; ++hd) {
    float v = asrc[sn * N_HEAD + hd] + adst[dn * N_HEAD + hd];
    v = (v > 0.f) ? v : 0.2f * v;                       // leaky_relu(0.2)
    atomicMaxF(&m[dn * N_HEAD + hd], v);
  }
}

// ---------------- pass 2: exp + per-(dst,head) sum ----------------
__global__ void k_edge_exp(const int* __restrict__ ei, const float* __restrict__ asrc,
                           const float* __restrict__ adst, const float* __restrict__ m,
                           float* __restrict__ w, float* __restrict__ s) {
  int e = blockIdx.x * blockDim.x + threadIdx.x;
  if (e >= ETOT) return;
  int sn, dn; edge_nodes(ei, e, sn, dn);
#pragma unroll
  for (int hd = 0; hd < N_HEAD; ++hd) {
    float v = asrc[sn * N_HEAD + hd] + adst[dn * N_HEAD + hd];
    v = (v > 0.f) ? v : 0.2f * v;
    float ex = __expf(v - m[dn * N_HEAD + hd]);
    w[(size_t)e * N_HEAD + hd] = ex;
    atomicAdd(&s[dn * N_HEAD + hd], ex);
  }
}

// ---------------- pass 3: num[dst] += e * h[src] (divide by s later) ----------------
__global__ void k_edge_agg(const int* __restrict__ ei, const float* __restrict__ h,
                           const float* __restrict__ w, float* __restrict__ num) {
  long long t = (long long)blockIdx.x * blockDim.x + threadIdx.x;
  int e = (int)(t >> 6);
  int c = (int)(t & 63);
  if (e >= ETOT) return;
  int sn, dn; edge_nodes(ei, e, sn, dn);
#pragma unroll
  for (int hd = 0; hd < N_HEAD; ++hd) {
    float coef = w[(size_t)e * N_HEAD + hd];
    float hv   = h[(size_t)sn * C_GAT + hd * C_HID + c];
    atomicAdd(&num[(size_t)dn * C_GAT + hd * C_HID + c], coef * hv);
  }
}

// ---------------- node finalize: head-mean(num/s)+bias -> BN -> ReLU ----------------
// do_pool==0: emit bf16 features for next GEMM; do_pool==1: atomically pool per graph
__global__ void k_node_final(const float* __restrict__ num, const float* __restrict__ s,
                             const float* __restrict__ bias, const float* __restrict__ g,
                             const float* __restrict__ b, const float* __restrict__ mn,
                             const float* __restrict__ vr, unsigned short* __restrict__ hnb,
                             float* __restrict__ pool_sum, float* __restrict__ pool_cnt,
                             const int* __restrict__ batch, int do_pool) {
  int t = blockIdx.x * blockDim.x + threadIdx.x;
  if (t >= N_NODES * C_HID) return;
  int c = t & (C_HID - 1);
  int n = t >> 6;
  float acc = 0.f;
#pragma unroll
  for (int hd = 0; hd < N_HEAD; ++hd)
    acc += num[(size_t)n * C_GAT + hd * C_HID + c] / s[n * N_HEAD + hd];
  acc = acc * 0.25f + bias[c];                                   // mean over heads + bias
  acc = (acc - mn[c]) * rsqrtf(vr[c] + BN_EPS) * g[c] + b[c];    // BN
  acc = fmaxf(acc, 0.f);                                         // ReLU
  if (do_pool) {
    int gi = batch[n];
    atomicAdd(&pool_sum[gi * C_HID + c], acc);
    if (c == 0) atomicAdd(&pool_cnt[gi], 1.0f);
  } else {
    hnb[t] = f2bf(acc);
  }
}

// ---------------- head: mean-pool -> fc1+ReLU -> fc2 ----------------
__global__ void k_head(const float* __restrict__ pool_sum, const float* __restrict__ pool_cnt,
                       const float* __restrict__ fc1w, const float* __restrict__ fc1b,
                       const float* __restrict__ fc2w, const float* __restrict__ fc2b,
                       float* __restrict__ out) {
  __shared__ float emb[C_HID];
  __shared__ float hid[C_OUT];
  int g = blockIdx.x, t = threadIdx.x;        // 64 threads per graph
  float cnt = fmaxf(pool_cnt[g], 1.0f);
  emb[t] = pool_sum[g * C_HID + t] / cnt;
  __syncthreads();
  if (t < C_OUT) {
    float a = fc1b[t];
#pragma unroll 8
    for (int c = 0; c < C_HID; ++c) a += emb[c] * fc1w[c * C_OUT + t];
    hid[t] = fmaxf(a, 0.f);
  }
  __syncthreads();
  if (t < C_OUT) {
    float a = fc2b[t];
#pragma unroll 8
    for (int j = 0; j < C_OUT; ++j) a += hid[j] * fc2w[j * C_OUT + t];
    out[g * C_OUT + t] = a;
  }
}

// ================= launch =================
static inline size_t alignup(size_t x) { return (x + 255) & ~(size_t)255; }

extern "C" void kernel_launch(void* const* d_in, const int* in_sizes, int n_in,
                              void* d_out, int out_size, void* d_ws, size_t ws_size,
                              hipStream_t stream) {
  const float* x       = (const float*)d_in[0];
  const int*   ei      = (const int*)d_in[1];
  const int*   batch   = (const int*)d_in[2];
  const float* in_g    = (const float*)d_in[3];
  const float* in_b    = (const float*)d_in[4];
  const float* in_m    = (const float*)d_in[5];
  const float* in_v    = (const float*)d_in[6];
  const float* W1      = (const float*)d_in[7];
  const float* as1     = (const float*)d_in[8];
  const float* ad1     = (const float*)d_in[9];
  const float* b1      = (const float*)d_in[10];
  const float* bn1g    = (const float*)d_in[11];
  const float* bn1b    = (const float*)d_in[12];
  const float* bn1m    = (const float*)d_in[13];
  const float* bn1v    = (const float*)d_in[14];
  const float* W2      = (const float*)d_in[15];
  const float* as2     = (const float*)d_in[16];
  const float* ad2     = (const float*)d_in[17];
  const float* b2      = (const float*)d_in[18];
  const float* bn2g    = (const float*)d_in[19];
  const float* bn2b    = (const float*)d_in[20];
  const float* bn2m    = (const float*)d_in[21];
  const float* bn2v    = (const float*)d_in[22];
  const float* fc1w    = (const float*)d_in[23];
  const float* fc1b    = (const float*)d_in[24];
  const float* fc2w    = (const float*)d_in[25];
  const float* fc2b    = (const float*)d_in[26];
  float* out = (float*)d_out;

  // workspace carve-up
  char* p = (char*)d_ws; size_t off = 0;
  unsigned short* xb   = (unsigned short*)(p + off); off = alignup(off + (size_t)N_NODES * C_IN * 2);
  unsigned short* W1p  = (unsigned short*)(p + off); off = alignup(off + (size_t)C_IN * C_GAT * 2);
  unsigned short* W2p  = (unsigned short*)(p + off); off = alignup(off + (size_t)C_HID * C_GAT * 2);
  float* h             = (float*)(p + off);          off = alignup(off + (size_t)N_NODES * C_GAT * 4);
  float* num           = (float*)(p + off);          off = alignup(off + (size_t)N_NODES * C_GAT * 4);
  float* asrc          = (float*)(p + off);          off = alignup(off + (size_t)N_NODES * N_HEAD * 4);
  float* adst          = (float*)(p + off);          off = alignup(off + (size_t)N_NODES * N_HEAD * 4);
  float* mbuf          = (float*)(p + off);          off = alignup(off + (size_t)N_NODES * N_HEAD * 4);
  float* sbuf          = (float*)(p + off);          off = alignup(off + (size_t)N_NODES * N_HEAD * 4);
  float* wbuf          = (float*)(p + off);          off = alignup(off + (size_t)ETOT * N_HEAD * 4);
  unsigned short* hnb  = (unsigned short*)(p + off); off = alignup(off + (size_t)N_NODES * C_HID * 2);
  float* pool_sum      = (float*)(p + off);          off = alignup(off + (size_t)N_GRAPH * C_HID * 4);
  float* pool_cnt      = (float*)(p + off);          off = alignup(off + (size_t)N_GRAPH * 4);
  (void)ws_size; (void)n_in; (void)in_sizes; (void)out_size;

  const int T = 256;
  auto blocks = [](long long n, int t) { return (unsigned)((n + t - 1) / t); };

  // ---- input BN -> bf16, pack weights ----
  k_bn_in<<<blocks((long long)N_NODES * C_IN, T), T, 0, stream>>>(x, in_g, in_b, in_m, in_v, xb);
  k_packW<<<blocks(C_IN * C_GAT, T), T, 0, stream>>>(W1, W1p, C_IN, C_GAT);
  k_packW<<<blocks(C_HID * C_GAT, T), T, 0, stream>>>(W2, W2p, C_HID, C_GAT);

  // ================= GAT layer 1 =================
  hipMemsetAsync(sbuf, 0, (size_t)N_NODES * N_HEAD * 4, stream);
  hipMemsetAsync(num, 0, (size_t)N_NODES * C_GAT * 4, stream);
  k_fill<<<blocks(N_NODES * N_HEAD, T), T, 0, stream>>>(mbuf, -3.0e38f, N_NODES * N_HEAD);

  {
    long long strips = (long long)(N_NODES / 16) * (C_GAT / 64);
    k_gemm_bf16<<<blocks(strips * 32, T), T, 0, stream>>>(xb, W1p, h, N_NODES, C_IN, C_GAT);
  }
  k_att<<<blocks(N_NODES * N_HEAD, T), T, 0, stream>>>(h, as1, ad1, asrc, adst);
  k_edge_max<<<blocks(ETOT, T), T, 0, stream>>>(ei, asrc, adst, mbuf);
  k_edge_exp<<<blocks(ETOT, T), T, 0, stream>>>(ei, asrc, adst, mbuf, wbuf, sbuf);
  k_edge_agg<<<blocks((long long)ETOT * C_HID, T), T, 0, stream>>>(ei, h, wbuf, num);
  k_node_final<<<blocks((long long)N_NODES * C_HID, T), T, 0, stream>>>(
      num, sbuf, b1, bn1g, bn1b, bn1m, bn1v, hnb, pool_sum, pool_cnt, batch, 0);

  // ================= GAT layer 2 =================
  hipMemsetAsync(sbuf, 0, (size_t)N_NODES * N_HEAD * 4, stream);
  hipMemsetAsync(num, 0, (size_t)N_NODES * C_GAT * 4, stream);
  hipMemsetAsync(pool_sum, 0, (size_t)N_GRAPH * C_HID * 4, stream);
  hipMemsetAsync(pool_cnt, 0, (size_t)N_GRAPH * 4, stream);
  k_fill<<<blocks(N_NODES * N_HEAD, T), T, 0, stream>>>(mbuf, -3.0e38f, N_NODES * N_HEAD);

  {
    long long strips = (long long)(N_NODES / 16) * (C_GAT / 64);
    k_gemm_bf16<<<blocks(strips * 32, T), T, 0, stream>>>(hnb, W2p, h, N_NODES, C_HID, C_GAT);
  }
  k_att<<<blocks(N_NODES * N_HEAD, T), T, 0, stream>>>(h, as2, ad2, asrc, adst);
  k_edge_max<<<blocks(ETOT, T), T, 0, stream>>>(ei, asrc, adst, mbuf);
  k_edge_exp<<<blocks(ETOT, T), T, 0, stream>>>(ei, asrc, adst, mbuf, wbuf, sbuf);
  k_edge_agg<<<blocks((long long)ETOT * C_HID, T), T, 0, stream>>>(ei, h, wbuf, num);
  k_node_final<<<blocks((long long)N_NODES * C_HID, T), T, 0, stream>>>(
      num, sbuf, b2, bn2g, bn2b, bn2m, bn2v, hnb, pool_sum, pool_cnt, batch, 1);

  // ================= head =================
  k_head<<<N_GRAPH, C_HID, 0, stream>>>(pool_sum, pool_cnt, fc1w, fc1b, fc2w, fc2b, out);
}
